// Decoder_21148418965753
// MI455X (gfx1250) — matrix operand order repaired and checked by reading, not compiled, not run on hardware
//
#include <hip/hip_runtime.h>

// ---------------------------------------------------------------------------
// PWC-Net decoder level for MI455X (gfx1250, wave32, WMMA, async-LDS).
//  * convs = implicit GEMM on v_wmma_f32_16x16x32_bf16 (f32 accumulate)
//  * bf16 NHWC activation mirror => conv staging is pure contiguous B128
//    GLOBAL_LOAD_ASYNC_TO_LDS copies (no f32->bf16 in the hot loop)
//  * double-buffered LDS, s_wait_asynccnt software pipeline
// ---------------------------------------------------------------------------

typedef __bf16 bf16_t;
typedef __attribute__((ext_vector_type(16))) __bf16 v16bf;
typedef __attribute__((ext_vector_type(8)))  __bf16 v8bf;
typedef __attribute__((ext_vector_type(8)))  float  v8f;
typedef int v4i_t __attribute__((vector_size(16)));   // builtin's 16B unit type

#define B_   16
#define H_   64
#define W_   128
#define HW_  (H_ * W_)     // 8192
#define FC_  269           // feat_out channels (NCHW f32 output)
#define CBF_ 304           // NHWC bf16 mirror channels (padded, [269,304) zero)
// feat_out channel slots (dense-net concat order after 4 convs):
//   [0:16)=c4  [16:40)=c3  [40:88)=c2  [88:136)=c1
//   [136:217)=corr  [217:265)=feat_1  [265:267)=upflow  [267:269)=upfeat
// Every conv input window ends exactly at channel 269 => zeroed pad channels
// provide the K-padding for all five convs.

#if __has_builtin(__builtin_amdgcn_global_load_async_to_lds_b128)
#define HAVE_ASYNC 1
#else
#define HAVE_ASYNC 0
#endif

#if __has_builtin(__builtin_amdgcn_s_wait_asynccnt)
#define WAIT_ASYNC(n) __builtin_amdgcn_s_wait_asynccnt(n)
#else
#define WAIT_ASYNC(n) asm volatile("s_wait_asynccnt %0" ::"i"(n))
#endif

static __device__ __forceinline__ void async_copy16(void* lds_dst, void* gsrc) {
#if HAVE_ASYNC
    __builtin_amdgcn_global_load_async_to_lds_b128(
        (__attribute__((address_space(1))) v4i_t*)gsrc,
        (__attribute__((address_space(3))) v4i_t*)lds_dst, 0, 0);
#else
    *(v8bf*)lds_dst = *(const v8bf*)gsrc;
#endif
}

static __device__ __forceinline__ bf16_t f2bf(float f) {
    unsigned int u = __builtin_bit_cast(unsigned int, f);
    unsigned int r = (u + 0x7FFFu + ((u >> 16) & 1u)) >> 16;
    unsigned short s = (unsigned short)r;
    return __builtin_bit_cast(bf16_t, s);
}

static __device__ __forceinline__ float leaky(float v) {
    return v >= 0.f ? v : 0.1f * v;
}

// ---------------------------------------------------------------------------
// Zero the mirror's pad channels [269, 304) once.
// ---------------------------------------------------------------------------
__global__ void zero_mirror_pad(bf16_t* __restrict__ featBf) {
    const int PAD = CBF_ - FC_;  // 35
    int idx = blockIdx.x * blockDim.x + threadIdx.x;
    if (idx >= B_ * HW_ * PAD) return;
    int ch = idx % PAD, pix = idx / PAD;
    featBf[(size_t)pix * CBF_ + FC_ + ch] = f2bf(0.f);
}

// ---------------------------------------------------------------------------
// Weight repack: f32 [Cout][Cin][3][3] ->
//   bf16 [chunk][(tap*CoutPad + n)*32 + cLocal], chunk stride = WCHUNK
// (exact LDS image per 32-channel chunk, zero padded to WCHUNK).
// ---------------------------------------------------------------------------
__global__ void prep_weights(const float* __restrict__ w, bf16_t* __restrict__ wb,
                             int Cin, int nCh, int Cout, int CoutPad, int wchunk) {
    int total = nCh * wchunk;
    for (int i = blockIdx.x * blockDim.x + threadIdx.x; i < total;
         i += gridDim.x * blockDim.x) {
        int chunk  = i / wchunk;
        int within = i % wchunk;
        int cl  = within & 31;
        int t2  = within >> 5;
        int n   = t2 % CoutPad;
        int tap = t2 / CoutPad;
        float v = 0.f;
        if (tap < 9 && n < Cout) {
            int c = chunk * 32 + cl;
            if (c < Cin) {
                int ky = tap / 3, kx = tap % 3;
                v = w[(((size_t)n * Cin + c) * 3 + ky) * 3 + kx];
            }
        }
        wb[i] = f2bf(v);
    }
}

// ---------------------------------------------------------------------------
// ConvTranspose2d(k=4,s=2,p=1), Cout=2: -> f32 slot + bf16 mirror (+ws flow).
// ---------------------------------------------------------------------------
__global__ void deconv4x4_s2_k(const float* __restrict__ in, int Cin,
                               const float* __restrict__ w,
                               const float* __restrict__ bias,
                               float* featOut, bf16_t* featBf, int chanBase,
                               float* wsDst) {
    const int H2 = H_ / 2, W2 = W_ / 2;
    int idx = blockIdx.x * blockDim.x + threadIdx.x;
    if (idx >= B_ * 2 * HW_) return;
    int p  = idx % HW_;
    int co = (idx / HW_) & 1;
    int b  = idx / (2 * HW_);
    int oy = p / W_, ox = p % W_;

    int kys[2], iys[2], nky = 0;
    int kxs[2], ixs[2], nkx = 0;
    for (int k = 0; k < 4; ++k) {
        int t = oy + 1 - k;
        if (t >= 0 && (t & 1) == 0 && (t >> 1) < H2) { kys[nky] = k; iys[nky] = t >> 1; ++nky; }
        int s = ox + 1 - k;
        if (s >= 0 && (s & 1) == 0 && (s >> 1) < W2) { kxs[nkx] = k; ixs[nkx] = s >> 1; ++nkx; }
    }
    float acc = bias[co];
    for (int ci = 0; ci < Cin; ++ci) {
        const float* ib = in + ((size_t)(b * Cin + ci) * H2) * W2;
        const float* wp = w + ((size_t)ci * 2 + co) * 16;
        for (int a = 0; a < nky; ++a)
            for (int q = 0; q < nkx; ++q)
                acc += ib[iys[a] * W2 + ixs[q]] * wp[kys[a] * 4 + kxs[q]];
    }
    featOut[((size_t)b * FC_ + chanBase + co) * HW_ + p] = acc;
    featBf[((size_t)b * HW_ + p) * CBF_ + chanBase + co] = f2bf(acc);
    if (wsDst) wsDst[((size_t)b * 2 + co) * HW_ + p] = acc;
}

// ---------------------------------------------------------------------------
// Backwarp feat_2 by 1.25*upped_flow (bilinear, zero pad, validity mask).
// ---------------------------------------------------------------------------
__global__ void backwarp_k(const float* __restrict__ feat2,
                           const float* __restrict__ uf,
                           float* __restrict__ warped) {
    int idx = blockIdx.x * blockDim.x + threadIdx.x;
    if (idx >= B_ * HW_) return;
    int b = idx / HW_, p = idx % HW_;
    int y = p / W_, x = p % W_;

    float xs = (float)x + uf[((size_t)b * 2 + 0) * HW_ + p] * (1.25f * (float)W_ / (W_ - 1));
    float ys = (float)y + uf[((size_t)b * 2 + 1) * HW_ + p] * (1.25f * (float)H_ / (H_ - 1));
    float x0f = floorf(xs), y0f = floorf(ys);
    float wx1 = xs - x0f, wx0 = 1.f - wx1;
    float wy1 = ys - y0f, wy0 = 1.f - wy1;
    int x0 = (int)x0f, y0 = (int)y0f, x1 = x0 + 1, y1 = y0 + 1;
    bool vx0 = (x0 >= 0) & (x0 < W_), vx1 = (x1 >= 0) & (x1 < W_);
    bool vy0 = (y0 >= 0) & (y0 < H_), vy1 = (y1 >= 0) & (y1 < H_);
    float wA = (vx0 && vy0) ? wx0 * wy0 : 0.f;
    float wB = (vx1 && vy0) ? wx1 * wy0 : 0.f;
    float wC = (vx0 && vy1) ? wx0 * wy1 : 0.f;
    float wD = (vx1 && vy1) ? wx1 * wy1 : 0.f;
    float keep = (wA + wB + wC + wD) > 0.999f ? 1.f : 0.f;
    int cx0 = min(max(x0, 0), W_ - 1), cx1 = min(max(x1, 0), W_ - 1);
    int cy0 = min(max(y0, 0), H_ - 1), cy1 = min(max(y1, 0), H_ - 1);

    const float* f2b = feat2 + (size_t)b * 48 * HW_;
    #pragma unroll 4
    for (int c = 0; c < 48; ++c) {
        const float* fc = f2b + (size_t)c * HW_;
        float v = wA * fc[cy0 * W_ + cx0] + wB * fc[cy0 * W_ + cx1]
                + wC * fc[cy1 * W_ + cx0] + wD * fc[cy1 * W_ + cx1];
        warped[((size_t)b * 48 + c) * HW_ + p] = v * keep;
    }
}

// ---------------------------------------------------------------------------
// 81-shift cost volume + leaky + feat_1 copy -> f32 slots + bf16 mirror.
// ---------------------------------------------------------------------------
__global__ void corr_k(const float* __restrict__ f1,
                       const float* __restrict__ warped,
                       float* __restrict__ featOut, bf16_t* __restrict__ featBf) {
    int idx = blockIdx.x * blockDim.x + threadIdx.x;
    if (idx >= B_ * HW_) return;
    int b = idx / HW_, p = idx % HW_;
    int y = p / W_, x = p % W_;

    float v[48];
    const float* f1b = f1 + (size_t)b * 48 * HW_ + p;
    #pragma unroll
    for (int c = 0; c < 48; ++c) v[c] = f1b[(size_t)c * HW_];

    float*  fo = featOut + (size_t)b * FC_ * HW_ + p;
    bf16_t* mb = featBf + ((size_t)b * HW_ + p) * CBF_;
    #pragma unroll
    for (int c = 0; c < 48; ++c) {
        fo[(size_t)(217 + c) * HW_] = v[c];
        mb[217 + c] = f2bf(v[c]);
    }

    const float* wbase = warped + (size_t)b * 48 * HW_;
    int k = 0;
    for (int dy = -4; dy <= 4; ++dy) {
        for (int dx = -4; dx <= 4; ++dx, ++k) {
            float s = 0.f;
            int yy = y + dy, xx = x + dx;
            if (yy >= 0 && yy < H_ && xx >= 0 && xx < W_) {
                const float* wp = wbase + yy * W_ + xx;
                #pragma unroll
                for (int c = 0; c < 48; ++c) s += v[c] * wp[(size_t)c * HW_];
            }
            float r = leaky(s * (1.f / 48.f));
            fo[(size_t)(136 + k) * HW_] = r;
            mb[136 + k] = f2bf(r);
        }
    }
}

// ---------------------------------------------------------------------------
// 3x3 conv as implicit GEMM on v_wmma_f32_16x16x32_bf16.
// Block = one (b,y) row, 8 waves x 16-pixel M tiles, NT x 16 out channels.
// K = 9*Cin as (tap x 32ch) chunks; double-buffered LDS fed by async B128
// copies from the NHWC bf16 mirror; s_wait_asynccnt pipeline.
// ---------------------------------------------------------------------------
template <int NT>
__global__ __launch_bounds__(256, 1)
void conv3x3_wmma(bf16_t* srcBf, float* dst, int dstTotC, int dstBase,
                  bf16_t* dstBf, const bf16_t* __restrict__ wPack,
                  const float* __restrict__ bias,
                  int nCh, int Cout, int srcBase) {
    constexpr int CoutPad = NT * 16;
    constexpr int UW      = (9 * CoutPad * 4 + 255) / 256;   // weight units/thread
    constexpr int WCHUNK  = UW * 256 * 8;                    // bf16 elems per chunk
    constexpr int S       = 6 + UW;                          // async ops per wave/stage

    __shared__ __attribute__((aligned(16))) bf16_t actLDS[2][3 * W_ * 32];
    __shared__ __attribute__((aligned(16))) bf16_t wLDS[2][WCHUNK];
    __shared__ __attribute__((aligned(16))) bf16_t trash[8];

    const int tid   = threadIdx.x;
    const int lane  = tid & 31;
    const int wave  = tid >> 5;
    const int khalf = lane >> 4;
    const int lm    = lane & 15;
    const int b     = blockIdx.x >> 6;   // grid = B*H
    const int y     = blockIdx.x & (H_ - 1);
    const int p0    = wave * 16;

    v8f acc[NT];
    #pragma unroll
    for (int nt = 0; nt < NT; ++nt)
        #pragma unroll
        for (int j = 0; j < 8; ++j) acc[nt][j] = 0.f;

    // Stage one 32-channel chunk into LDS buffer `buf` (uniform S async ops
    // per wave: OOB halo rows redirect to a trash slot + synchronous zero).
    auto stage = [&](int chunk, int buf) {
        const int c0 = chunk * 32;
        #pragma unroll
        for (int k = 0; k < 6; ++k) {         // 1536 units = 6 per thread, exact
            int i = tid + k * 256;
            int q = i & 3, x = (i >> 2) & 127, r = i >> 9;
            int yy = y + r - 1;
            bf16_t* ldst;
            bf16_t* gsrc;
            if (yy >= 0 && yy < H_) {
                ldst = &actLDS[buf][(r * W_ + x) * 32 + q * 8];
                gsrc = srcBf + ((size_t)(b * H_ + yy) * W_ + x) * CBF_
                             + srcBase + c0 + q * 8;
            } else {
                v8bf z;
                #pragma unroll
                for (int j = 0; j < 8; ++j) z[j] = f2bf(0.f);
                *(v8bf*)&actLDS[buf][(r * W_ + x) * 32 + q * 8] = z;
                ldst = trash;
                gsrc = srcBf;
            }
            async_copy16(ldst, gsrc);
        }
        const bf16_t* wc = wPack + (size_t)chunk * WCHUNK;
        #pragma unroll
        for (int u = 0; u < UW; ++u) {
            int e = (tid + u * 256) * 8;
            async_copy16(&wLDS[buf][e], (void*)(wc + e));
        }
    };

    stage(0, 0);
    for (int ci = 0; ci < nCh; ++ci) {
        const int buf = ci & 1;
        if (ci + 1 < nCh) {
            stage(ci + 1, buf ^ 1);
            WAIT_ASYNC(S);      // chunk ci's copies retired (in-order per wave)
        } else {
            WAIT_ASYNC(0);
        }
        __syncthreads();

        #pragma unroll
        for (int tap = 0; tap < 9; ++tap) {
            const int ky = tap / 3, kx = tap % 3;
            union { v16bf v; v8bf h[2]; } A;
            const int xs = p0 + lm + kx - 1;
            if (xs >= 0 && xs < W_) {
                const int abase = (ky * W_ + xs) * 32;
                A.h[0] = *reinterpret_cast<const v8bf*>(&actLDS[buf][abase + khalf * 8]);
                A.h[1] = *reinterpret_cast<const v8bf*>(&actLDS[buf][abase + 16 + khalf * 8]);
            } else {
                #pragma unroll
                for (int j = 0; j < 16; ++j) A.v[j] = f2bf(0.f);
            }
            #pragma unroll
            for (int nt = 0; nt < NT; ++nt) {
                const int n = nt * 16 + lm;
                const int bbase = (tap * CoutPad + n) * 32 + khalf * 8;
                union { v16bf v; v8bf h[2]; } Bm;
                Bm.h[0] = *reinterpret_cast<const v8bf*>(&wLDS[buf][bbase]);
                Bm.h[1] = *reinterpret_cast<const v8bf*>(&wLDS[buf][bbase + 16]);
                acc[nt] = __builtin_amdgcn_wmma_f32_16x16x32_bf16(
                    false, A.v, false, Bm.v, (short)0, acc[nt], false, false);
            }
        }
        __syncthreads();
    }

    // Epilogue: D layout -> lane (l&15)=N, VGPR v => pixel p0 + 8*khalf + v.
    #pragma unroll
    for (int nt = 0; nt < NT; ++nt) {
        const int cout = nt * 16 + lm;
        if (cout < Cout) {
            const float bv = bias[cout];
            float* dp = dst + ((size_t)b * dstTotC + dstBase + cout) * HW_
                            + (size_t)y * W_ + p0 + khalf * 8;
            const size_t pixBase = ((size_t)b * H_ + y) * W_ + p0 + khalf * 8;
            #pragma unroll
            for (int v = 0; v < 8; ++v) {
                float val = leaky(acc[nt][v] + bv);
                dp[v] = val;
                if (dstBf)
                    dstBf[(pixBase + v) * CBF_ + dstBase + cout] = f2bf(val);
            }
        }
    }
}

// ---------------------------------------------------------------------------
extern "C" void kernel_launch(void* const* d_in, const int* in_sizes, int n_in,
                              void* d_out, int out_size, void* d_ws, size_t ws_size,
                              hipStream_t stream) {
    (void)in_sizes; (void)n_in; (void)out_size; (void)ws_size;

    const float* feat1     = (const float*)d_in[0];
    const float* feat2     = (const float*)d_in[1];
    const float* prev_flow = (const float*)d_in[2];
    const float* prev_feat = (const float*)d_in[3];
    const float* w1 = (const float*)d_in[4];   const float* b1 = (const float*)d_in[5];
    const float* w2 = (const float*)d_in[6];   const float* b2 = (const float*)d_in[7];
    const float* w3 = (const float*)d_in[8];   const float* b3 = (const float*)d_in[9];
    const float* w4 = (const float*)d_in[10];  const float* b4 = (const float*)d_in[11];
    const float* w5 = (const float*)d_in[12];  const float* b5 = (const float*)d_in[13];
    const float* w_upflow = (const float*)d_in[14]; const float* b_upflow = (const float*)d_in[15];
    const float* w_upfeat = (const float*)d_in[16]; const float* b_upfeat = (const float*)d_in[17];

    float* flowOut = (float*)d_out;                       // [16,2,64,128]
    float* featOut = flowOut + (size_t)B_ * 2 * HW_;      // [16,269,64,128]

    // Workspace carve-up.
    char* wsp = (char*)d_ws;
    size_t off = 0;
    auto carve = [&](size_t bytes) {
        void* p = wsp + off;
        off = (off + bytes + 255) & ~(size_t)255;
        return p;
    };
    bf16_t* featBf = (bf16_t*)carve((size_t)B_ * HW_ * CBF_ * sizeof(bf16_t));
    float*  warped = (float*)carve((size_t)B_ * 48 * HW_ * 4);
    float*  upflow = (float*)carve((size_t)B_ * 2 * HW_ * 4);

    const int cin[5]   = {133, 181, 229, 253, 269};
    const int cout[5]  = {48, 48, 24, 16, 2};
    const int nCh[5]   = {5, 6, 8, 8, 9};        // ceil(Cin/32)
    const int coutP[5] = {48, 48, 32, 16, 16};
    const int uw[5]    = {7, 7, 5, 3, 3};        // (9*CoutPad*4 + 255)/256
    bf16_t* wbk[5];
    int wchunk[5];
    for (int k = 0; k < 5; ++k) {
        wchunk[k] = uw[k] * 256 * 8;
        wbk[k] = (bf16_t*)carve((size_t)nCh[k] * wchunk[k] * sizeof(bf16_t));
    }

    // 0) Zero mirror pad channels.
    zero_mirror_pad<<<(B_ * HW_ * (CBF_ - FC_) + 255) / 256, 256, 0, stream>>>(featBf);

    // 1) Weight repack to per-chunk bf16 LDS images.
    const float* wsrc[5] = {w1, w2, w3, w4, w5};
    for (int k = 0; k < 5; ++k) {
        int total = nCh[k] * wchunk[k];
        prep_weights<<<(total + 255) / 256, 256, 0, stream>>>(
            wsrc[k], wbk[k], cin[k], nCh[k], cout[k], coutP[k], wchunk[k]);
    }

    // 2) Deconvs -> feat slots + mirror (+ f32 upflow for backwarp).
    deconv4x4_s2_k<<<(B_ * 2 * HW_ + 255) / 256, 256, 0, stream>>>(
        prev_flow, 2, w_upflow, b_upflow, featOut, featBf, 265, upflow);
    deconv4x4_s2_k<<<(B_ * 2 * HW_ + 255) / 256, 256, 0, stream>>>(
        prev_feat, 217, w_upfeat, b_upfeat, featOut, featBf, 267, nullptr);

    // 3) Backwarp feat_2 by 1.25 * upped_flow.
    backwarp_k<<<(B_ * HW_ + 255) / 256, 256, 0, stream>>>(feat2, upflow, warped);

    // 4) Cost volume + feat_1 copy -> slots + mirror.
    corr_k<<<(B_ * HW_ + 255) / 256, 256, 0, stream>>>(feat1, warped, featOut, featBf);

    // 5) Dense-block convs (WMMA, async-LDS pipeline). conv5 -> flow.
    dim3 cgrid(B_ * H_);
    conv3x3_wmma<3><<<cgrid, 256, 0, stream>>>(featBf, featOut, FC_, 88, featBf,
                                               wbk[0], b1, nCh[0], 48, 136);
    conv3x3_wmma<3><<<cgrid, 256, 0, stream>>>(featBf, featOut, FC_, 40, featBf,
                                               wbk[1], b2, nCh[1], 48, 88);
    conv3x3_wmma<2><<<cgrid, 256, 0, stream>>>(featBf, featOut, FC_, 16, featBf,
                                               wbk[2], b3, nCh[2], 24, 40);
    conv3x3_wmma<1><<<cgrid, 256, 0, stream>>>(featBf, featOut, FC_, 0, featBf,
                                               wbk[3], b4, nCh[3], 16, 16);
    conv3x3_wmma<1><<<cgrid, 256, 0, stream>>>(featBf, flowOut, 2, 0, nullptr,
                                               wbk[4], b5, nCh[4], 2, 0);
}
